// StyleBlock_65343632441353
// MI455X (gfx1250) — matrix operand order repaired and checked by reading, not compiled, not run on hardware
//
#include <hip/hip_runtime.h>
#include <hip/hip_bf16.h>

// ---------------------------------------------------------------------------
// StyleGAN2 style-block on gfx1250: bf16 WMMA implicit GEMM + TDM staging.
// ---------------------------------------------------------------------------

typedef __bf16 bf16_t;
typedef __attribute__((ext_vector_type(16))) __bf16        v16bf;
typedef __attribute__((ext_vector_type(8)))  float         v8f;
typedef __attribute__((ext_vector_type(4)))  unsigned int  u32x4;  // trivial, union-safe
typedef __attribute__((ext_vector_type(8)))  int           i32x8;
typedef __attribute__((ext_vector_type(4)))  int           i32x4;

union BfVec { v16bf v; u32x4 q[2]; };

#define BATCH 16
#define CHAN  256
#define SDIM  512
#define HH    64
#define WW    64
#define HWPX  (HH * WW)

#if __has_builtin(__builtin_amdgcn_tensor_load_to_lds) && \
    __has_builtin(__builtin_amdgcn_s_wait_tensorcnt)
#define USE_TDM 1
#else
#define USE_TDM 0
#endif

// ---------------------------------------------------------------------------
// 1) NCHW f32 -> NHWC bf16 transpose (LDS-tiled, coalesced both sides)
// ---------------------------------------------------------------------------
__global__ __launch_bounds__(256)
void nchw2nhwc_bf16(const float* __restrict__ in, bf16_t* __restrict__ out) {
  const int pxb = blockIdx.x * 64;
  const int c0  = blockIdx.y * 32;
  const int b   = blockIdx.z;
  __shared__ float tile[32][65];
  const int t = threadIdx.x;
#pragma unroll
  for (int i = 0; i < 8; ++i) {
    int r = t + 256 * i;
    int cl = r >> 6, pl = r & 63;
    tile[cl][pl] = in[(size_t)(b * CHAN + c0 + cl) * HWPX + pxb + pl];
  }
  __syncthreads();
#pragma unroll
  for (int i = 0; i < 8; ++i) {
    int w = t + 256 * i;
    int cl = w & 31, pl = w >> 5;
    out[((size_t)b * HWPX + pxb + pl) * CHAN + c0 + cl] = (bf16_t)tile[cl][pl];
  }
}

// ---------------------------------------------------------------------------
// 2) style[b][c] = (wvec[b]*scale) . sw[c] + sb[c]
// ---------------------------------------------------------------------------
__global__ __launch_bounds__(256)
void style_kernel(const float* __restrict__ wvec, const float* __restrict__ sw,
                  const float* __restrict__ sb, float* __restrict__ style) {
  const int idx = blockIdx.x * blockDim.x + threadIdx.x;  // b*C + c
  const int b = idx / CHAN, c = idx % CHAN;
  float acc = 0.f;
  for (int s = 0; s < SDIM; ++s) acc += wvec[b * SDIM + s] * sw[c * SDIM + s];
  style[idx] = acc * (1.0f / 32.0f) + sb[c];  // GAIN * S^-0.5 = 1/32
}

// ---------------------------------------------------------------------------
// 3) Modulate + demodulate, emit bf16 weights as [B][tap][cout][cin]
// ---------------------------------------------------------------------------
__global__ __launch_bounds__(256)
void modw_kernel(const float* __restrict__ w, const float* __restrict__ style,
                 bf16_t* __restrict__ wmod) {
  const int cout = blockIdx.x & (CHAN - 1);
  const int b    = blockIdx.x >> 8;
  const int cin  = threadIdx.x;
  const float st = style[b * CHAN + cin];
  float wl[9];
  float ss = 0.f;
#pragma unroll
  for (int t = 0; t < 9; ++t) {
    float v = w[((size_t)cout * CHAN + cin) * 9 + t] * st;
    wl[t] = v;
    ss += v * v;
  }
  __shared__ float red[256];
  __shared__ float demod_s;
  red[cin] = ss;
  __syncthreads();
  for (int s = 128; s > 0; s >>= 1) {
    if (cin < s) red[cin] += red[cin + s];
    __syncthreads();
  }
  if (cin == 0) demod_s = rsqrtf(red[0] + 1e-8f);
  __syncthreads();
  const float demod = demod_s;
#pragma unroll
  for (int t = 0; t < 9; ++t)
    wmod[(((size_t)(b * 9 + t) * CHAN + cout) * CHAN) + cin] = (bf16_t)(wl[t] * demod);
}

// ---------------------------------------------------------------------------
// 4) Implicit-GEMM 3x3 conv. Block = (b, 64-cout tile, one 64-pixel row y).
//    All 3 halo rows staged in LDS once (TDM async copy for interior rows),
//    then the full K loop (3 dy x 8 cin-chunks x 3 dx) runs barrier-free:
//    144 v_wmma_f32_16x16x32_bf16 per wave. Epilogue fuses noise + lrelu.
// ---------------------------------------------------------------------------
template <bool WRITE_F32_NCHW>
__global__ __launch_bounds__(256)
void modconv_wmma(const bf16_t* __restrict__ in_nhwc,   // [B][H][W][C] bf16
                  const bf16_t* __restrict__ wmod,      // [B][9][Cout][Cin] bf16
                  const float*  __restrict__ noise,     // [B][H*W]
                  const float*  __restrict__ nw,        // [Cout]
                  bf16_t* __restrict__ out_nhwc,        // pass1 output
                  float*  __restrict__ out_nchw) {      // pass2 output
  const int y  = blockIdx.x;        // 0..63
  const int cb = blockIdx.y;        // 0..3  (cout block of 64)
  const int b  = blockIdx.z;        // 0..15

  const int tid   = threadIdx.x;
  const int lane  = tid & 31;
  const int wid   = tid >> 5;       // 0..7
  const int waveM = wid & 3;        // 0..3
  const int waveN = wid >> 2;       // 0..1
  const int half  = lane >> 4;      // K-half select
  const int l15   = lane & 15;

  // 3 halo rows x 66 px x 256 cin, bf16 = 101376 B; + epilogue transpose tile.
  __shared__ __align__(128) bf16_t lbuf3[3 * 66 * CHAN];
  __shared__ __align__(32)  bf16_t otile[64][66];

  const u32x4 zq = {0, 0, 0, 0};

  // ---- zero the constant x-halo columns (px 0 and 65 of each row) ----
  if (tid < 192) {                          // 3 rows * 2 cols * 512B = 192*16B
    const int r = tid >> 6, rem = tid & 63;
    const int s = rem >> 5, i = rem & 31;
    *(u32x4*)((char*)lbuf3 + (size_t)((r * 66 + s * 65) * CHAN) * 2 + i * 16) = zq;
  }

  // ---- stage the 3 input rows (interior 64 px, contiguous 32KB each) ----
#if USE_TDM
  bool issued_tdm = false;
#endif
  for (int dy = 0; dy < 3; ++dy) {
    const int yy = y + dy - 1;
    bf16_t* ldsRow = lbuf3 + (dy * 66 + 1) * CHAN;
    if (yy >= 0 && yy < HH) {
      const bf16_t* gRow = in_nhwc + (size_t)(b * HH + yy) * WW * CHAN;
#if USE_TDM
      if (wid == 0) {
        // Tensor DMA descriptor (ISA 8.3/8.4): 1-D tile, data_size=8B,
        // tile_dim0 = tensor_dim0 = 32768B/8 = 4096 units, type=2 (image).
        const unsigned long long ga = (unsigned long long)(uintptr_t)gRow;
        const unsigned int la = (unsigned int)(uintptr_t)ldsRow;  // LDS offset
        u32x4 g0;
        g0[0] = 1u;                                        // count=1, user mode
        g0[1] = la;                                        // lds_addr
        g0[2] = (unsigned int)(ga & 0xFFFFFFFFu);          // global_addr[31:0]
        g0[3] = (unsigned int)((ga >> 32) & 0x01FFFFFFu)   // global_addr[56:32]
                | 0x80000000u;                             // type=2 at [127:126]
        i32x8 g1;
        g1[0] = 0x00030000;           // workgroup_mask=0, data_size=3 (8B)
        g1[1] = (int)(4096u << 16);   // tensor_dim0[15:0] -> bits[31:16]
        g1[2] = (int)(1u << 16);      // tensor_dim0[31:16]=0, tensor_dim1=1
        g1[3] = (int)(4096u << 16);   // tensor_dim1 hi=0, tile_dim0=4096
        g1[4] = 1;                    // tile_dim1=1, tile_dim2=0
        g1[5] = 4096;                 // tensor_dim0_stride[31:0]
        g1[6] = 0;                    // stride hi / tensor_dim1_stride lo
        g1[7] = 0;                    // tensor_dim1_stride hi
        i32x4 g2 = {0, 0, 0, 0};
        i32x4 g3 = {0, 0, 0, 0};
        i32x8 g4 = {0, 0, 0, 0, 0, 0, 0, 0};   // 6-arg toolchain: extra group
        __builtin_amdgcn_tensor_load_to_lds(g0, g1, g2, g3, g4, 0);
        issued_tdm = true;
      }
#else
#pragma unroll
      for (int i = 0; i < 8; ++i)   // 2048 u32x4 / 256 threads, coalesced
        ((u32x4*)ldsRow)[tid + 256 * i] = ((const u32x4*)gRow)[tid + 256 * i];
#endif
    } else {
#pragma unroll
      for (int i = 0; i < 8; ++i)
        ((u32x4*)ldsRow)[tid + 256 * i] = zq;
    }
  }
#if USE_TDM
  if (issued_tdm) __builtin_amdgcn_s_wait_tensorcnt(0);
#endif
  __syncthreads();

  // ---- barrier-free K loop: 72 K-steps x (1 A-frag, 2 B-frags, 2 WMMA) ----
  v8f acc0 = {};
  v8f acc1 = {};
  const int coutRow = cb * 64 + waveM * 16 + l15;  // A row owned by this lane
  const int nPix0   = waveN * 32 + l15;            // B column pixel (acc0)

  for (int dy = 0; dy < 3; ++dy) {
    for (int chunk = 0; chunk < 8; ++chunk) {
#pragma unroll
      for (int dx = 0; dx < 3; ++dx) {
        const int tap = dy * 3 + dx;
        // A fragment: per-lane K run [16*half, +16)
        const bf16_t* arow = wmod +
            (((size_t)(b * 9 + tap) * CHAN + coutRow) * CHAN + chunk * 32);
        BfVec a;
        a.q[0] = *(const u32x4*)(arow + half * 16);
        a.q[1] = *(const u32x4*)(arow + half * 16 + 8);
        // B fragments: 16 consecutive cin for this lane's pixel column
        const bf16_t* brow = lbuf3 + (dy * 66) * CHAN + chunk * 32 + half * 16;
        const v16bf b0 = *(const v16bf*)(brow + (nPix0 + dx) * CHAN);
        const v16bf b1 = *(const v16bf*)(brow + (nPix0 + 16 + dx) * CHAN);
        acc0 = __builtin_amdgcn_wmma_f32_16x16x32_bf16(
            false, a.v, false, b0, (short)0, acc0, false, false);
        acc1 = __builtin_amdgcn_wmma_f32_16x16x32_bf16(
            false, a.v, false, b1, (short)0, acc1, false, false);
      }
    }
  }

  // ---- fused epilogue: + nw[cout]*noise, leaky_relu(0.2) ----
#pragma unroll
  for (int r = 0; r < 8; ++r) {
    const int m    = r + 8 * half;               // C-layout: M = r + 8*(lane>=16)
    const int cout = cb * 64 + waveM * 16 + m;
    const float nwv = nw[cout];
#pragma unroll
    for (int t = 0; t < 2; ++t) {
      const int x = nPix0 + t * 16;
      float v = (t == 0) ? acc0[r] : acc1[r];
      v += nwv * noise[b * HWPX + y * WW + x];
      v = v > 0.f ? v : 0.2f * v;
      if (WRITE_F32_NCHW) {
        out_nchw[(((size_t)b * CHAN + cout) * HH + y) * WW + x] = v;
      } else {
        otile[waveM * 16 + m][x] = (bf16_t)v;    // stage for coalesced NHWC store
      }
    }
  }
  if (!WRITE_F32_NCHW) {
    __syncthreads();
#pragma unroll
    for (int i = 0; i < 16; ++i) {
      int w  = tid + 256 * i;                    // 0..4095
      int cl = w & 63, pl = w >> 6;
      out_nhwc[(((size_t)b * HH + y) * WW + pl) * CHAN + cb * 64 + cl] =
          otile[cl][pl];
    }
  }
}

// ---------------------------------------------------------------------------
// Host-side orchestration (all on `stream`; workspace carved from d_ws)
// ---------------------------------------------------------------------------
extern "C" void kernel_launch(void* const* d_in, const int* in_sizes, int n_in,
                              void* d_out, int out_size, void* d_ws, size_t ws_size,
                              hipStream_t stream) {
  const float* x       = (const float*)d_in[0];   // [16,256,64,64]
  const float* w1      = (const float*)d_in[1];   // [16,512]
  const float* w2      = (const float*)d_in[2];   // [16,512]
  const float* noise1  = (const float*)d_in[3];   // [16,1,64,64]
  const float* noise2  = (const float*)d_in[4];   // [16,1,64,64]
  const float* s1_w    = (const float*)d_in[5];   // [256,512]
  const float* s1_b    = (const float*)d_in[6];   // [256]
  const float* conv1_w = (const float*)d_in[7];   // [256,256,3,3]
  const float* nw1     = (const float*)d_in[8];   // [256]
  const float* s2_w    = (const float*)d_in[9];   // [256,512]
  const float* s2_b    = (const float*)d_in[10];  // [256]
  const float* conv2_w = (const float*)d_in[11];  // [256,256,3,3]
  const float* nw2     = (const float*)d_in[12];  // [256]
  float* out = (float*)d_out;                     // [16,256,64,64] f32

  char* ws = (char*)d_ws;
  const size_t WMOD_BYTES = (size_t)BATCH * 9 * CHAN * CHAN * sizeof(bf16_t); // 18.87 MB
  const size_t ACT_BYTES  = (size_t)BATCH * HWPX * CHAN * sizeof(bf16_t);     // 33.55 MB
  float*  style1 = (float*)(ws + 0);
  float*  style2 = (float*)(ws + 16384);
  bf16_t* wmod1  = (bf16_t*)(ws + 32768);
  bf16_t* wmod2  = (bf16_t*)(ws + 32768 + WMOD_BYTES);
  bf16_t* xnhwc  = (bf16_t*)(ws + 32768 + 2 * WMOD_BYTES);
  bf16_t* hnhwc  = (bf16_t*)(ws + 32768 + 2 * WMOD_BYTES + ACT_BYTES);

  // 1) input layout conversion: NCHW f32 -> NHWC bf16
  nchw2nhwc_bf16<<<dim3(64, 8, 16), 256, 0, stream>>>(x, xnhwc);
  // 2) style vectors
  style_kernel<<<16, 256, 0, stream>>>(w1, s1_w, s1_b, style1);
  style_kernel<<<16, 256, 0, stream>>>(w2, s2_w, s2_b, style2);
  // 3) modulated + demodulated bf16 weights, tap-major layout
  modw_kernel<<<BATCH * CHAN, 256, 0, stream>>>(conv1_w, style1, wmod1);
  modw_kernel<<<BATCH * CHAN, 256, 0, stream>>>(conv2_w, style2, wmod2);
  // 4) conv1 (writes bf16 NHWC h) then conv2 (writes f32 NCHW out)
  modconv_wmma<false><<<dim3(64, 4, 16), 256, 0, stream>>>(
      xnhwc, wmod1, noise1, nw1, hnhwc, nullptr);
  modconv_wmma<true><<<dim3(64, 4, 16), 256, 0, stream>>>(
      hnhwc, wmod2, noise2, nw2, nullptr, out);
}
